// Attention_86818468921790
// MI455X (gfx1250) — matrix operand order repaired and checked by reading, not compiled, not run on hardware
//
#include <hip/hip_runtime.h>
#include <math.h>

// ---------------------------------------------------------------------------
// Types for CDNA5 WMMA (wave32): D(16x16 f32) = A(16x32 f16) * B(32x16 f16) + C
// ---------------------------------------------------------------------------
typedef __attribute__((ext_vector_type(16))) _Float16 v16h;
typedef __attribute__((ext_vector_type(8)))  _Float16 v8h;
typedef __attribute__((ext_vector_type(4)))  _Float16 v4h;
typedef __attribute__((ext_vector_type(8)))  float    v8f;

constexpr int kDim   = 2048;
constexpr int kSeq   = 2048;
constexpr int kBsz   = 2;
constexpr int kNH    = 32;
constexpr int kNKV   = 8;
constexpr int kHD    = 64;
constexpr int kKVDim = kNKV * kHD;   // 512
constexpr int kM     = kBsz * kSeq;  // 4096 rows for all GEMMs

__device__ __forceinline__ v8f wmma32(v16h a, v16h b, v8f c) {
  return __builtin_amdgcn_wmma_f32_16x16x32_f16(false, a, false, b, (short)0, c,
                                                false, false);
}

__device__ __forceinline__ v16h cat8(v8h lo, v8h hi) {
  return __builtin_shufflevector(lo, hi, 0, 1, 2, 3, 4, 5, 6, 7, 8, 9, 10, 11,
                                 12, 13, 14, 15);
}

// ---------------------------------------------------------------------------
// One-time precision conversion: f32 -> f16 (element-wise, float4 granularity)
// ---------------------------------------------------------------------------
__global__ __launch_bounds__(256)
void cvt_f32_f16(const float* __restrict__ in, _Float16* __restrict__ out,
                 int n4) {
  int i = blockIdx.x * 256 + threadIdx.x;
  if (i >= n4) return;
  float4 v = reinterpret_cast<const float4*>(in)[i];
  v4h o = {(_Float16)v.x, (_Float16)v.y, (_Float16)v.z, (_Float16)v.w};
  reinterpret_cast<v4h*>(out)[i] = o;
}

// ---------------------------------------------------------------------------
// One-time weight relayout: f32 [K][N] -> f16 [N][K] (LDS-tiled transpose so
// both the global read and the global write stay coalesced)
// ---------------------------------------------------------------------------
__global__ __launch_bounds__(256)
void transpose_f32_f16(const float* __restrict__ in, _Float16* __restrict__ out,
                       int K, int N) {
  __shared__ _Float16 tile[32][33];
  int kb = blockIdx.y * 32, nb = blockIdx.x * 32;
  int tx = threadIdx.x, ty = threadIdx.y;  // (32, 8)
#pragma unroll
  for (int i = 0; i < 4; ++i) {
    int r = ty + i * 8;
    tile[r][tx] = (_Float16)in[(size_t)(kb + r) * N + nb + tx];
  }
  __syncthreads();
#pragma unroll
  for (int i = 0; i < 4; ++i) {
    int r = ty + i * 8;
    out[(size_t)(nb + r) * K + kb + tx] = tile[tx][r];
  }
}

// ---------------------------------------------------------------------------
// GEMM: C[M x N](f32) = A[M x K](f16, row-major) * BT[N x K](f16, col-major B).
// No LDS: both operand layouts are fragment-native, so every fragment is two
// contiguous global b128 loads; cross-wave reuse comes from L0/L2 (operands
// are fully L2-resident on MI455X's 192MB L2).
// Block = 4 waves; block tile 128x64; each wave a 32x64 (2x4 WMMA) subtile.
// ---------------------------------------------------------------------------
__global__ __launch_bounds__(128)
void gemm_wmma_f16(const _Float16* __restrict__ A,
                   const _Float16* __restrict__ BT, float* __restrict__ C,
                   int M, int N, int K) {
  (void)M;
  const int t    = threadIdx.x;
  const int lane = t & 31;
  const int w    = t >> 5;
  const int m0   = blockIdx.y * 128 + w * 32;
  const int n0   = blockIdx.x * 64;
  const int row  = lane & 15;
  const int kh   = lane >> 4;

  const _Float16* aRow[2] = {A + (size_t)(m0 + row) * K,
                             A + (size_t)(m0 + 16 + row) * K};
  const _Float16* bCol[4];
#pragma unroll
  for (int j = 0; j < 4; ++j)
    bCol[j] = BT + (size_t)(n0 + j * 16 + row) * K;

  v8f acc[2][4] = {};

  for (int k0 = 0; k0 < K; k0 += 32) {
    // Prefetch next K-step (global_prefetch_b8); no LDS pipeline to hide
    // latency otherwise.
    if (k0 + 32 < K) {
      __builtin_prefetch(aRow[0] + k0 + 32, 0, 1);
      __builtin_prefetch(aRow[1] + k0 + 32, 0, 1);
      __builtin_prefetch(bCol[0] + k0 + 32, 0, 1);
      __builtin_prefetch(bCol[2] + k0 + 32, 0, 1);
    }
    v16h aF[2], bF[4];
#pragma unroll
    for (int i = 0; i < 2; ++i) {
      // A 16x32 frag: lane row = lane&15; element e <-> K=16*kh+e
      // (two b128 loads)
      v8h lo = *reinterpret_cast<const v8h*>(aRow[i] + k0 + kh * 16);
      v8h hi = *reinterpret_cast<const v8h*>(aRow[i] + k0 + kh * 16 + 8);
      aF[i] = cat8(lo, hi);
    }
#pragma unroll
    for (int j = 0; j < 4; ++j) {
      // B 32x16 frag: lane col = lane&15; element e <-> K=16*kh+e
      // (16 contiguous halves = two b128 loads)
      const _Float16* p = bCol[j] + k0 + kh * 16;
      bF[j] = cat8(*reinterpret_cast<const v8h*>(p),
                   *reinterpret_cast<const v8h*>(p + 8));
    }
#pragma unroll
    for (int i = 0; i < 2; ++i)
#pragma unroll
      for (int j = 0; j < 4; ++j)
        acc[i][j] = wmma32(aF[i], bF[j], acc[i][j]);
  }

  // C layout: VGPR r -> M = r + 8*(lane>>4), N = lane&15
#pragma unroll
  for (int i = 0; i < 2; ++i)
#pragma unroll
    for (int j = 0; j < 4; ++j) {
      int col   = n0 + j * 16 + row;
      int rbase = m0 + i * 16 + kh * 8;
#pragma unroll
      for (int r = 0; r < 8; ++r)
        C[(size_t)(rbase + r) * N + col] = acc[i][j][r];
    }
}

// ---------------------------------------------------------------------------
// RoPE + relayout + f16 down-convert:
//   qf [b][s][h*64+d]   -> qr16 [b][h][s][d]
//   kf [b][s][kvh*64+d] -> kr16 [b][kvh][s][d]
//   vf [b][s][kvh*64+d] -> vt16 [b][kvh][d][s]   (transposed for PV B-frags)
// ---------------------------------------------------------------------------
__global__ __launch_bounds__(256)
void rope_qkv(const float* __restrict__ qf, const float* __restrict__ kf,
              const float* __restrict__ vf, const float* __restrict__ cosT,
              const float* __restrict__ sinT, _Float16* __restrict__ qr,
              _Float16* __restrict__ kr, _Float16* __restrict__ vt) {
  int idx = blockIdx.x * blockDim.x + threadIdx.x;  // [b][s][h][j], 2^22 total
  int j = idx & 31;
  int h = (idx >> 5) & 31;
  int s = (idx >> 10) & (kSeq - 1);
  int b = idx >> 21;
  float c  = cosT[s * 32 + j];
  float sn = sinT[s * 32 + j];
  {
    const float* p = qf + ((size_t)(b * kSeq + s)) * kDim + h * kHD + 2 * j;
    float tr = p[0], ti = p[1];
    _Float16* o = qr + (((size_t)(b * kNH + h)) * kSeq + s) * kHD + 2 * j;
    o[0] = (_Float16)(tr * c - ti * sn);
    o[1] = (_Float16)(tr * sn + ti * c);
  }
  if (h < kNKV) {
    const float* p = kf + ((size_t)(b * kSeq + s)) * kKVDim + h * kHD + 2 * j;
    float tr = p[0], ti = p[1];
    _Float16* o = kr + (((size_t)(b * kNKV + h)) * kSeq + s) * kHD + 2 * j;
    o[0] = (_Float16)(tr * c - ti * sn);
    o[1] = (_Float16)(tr * sn + ti * c);
    const float* pv = vf + ((size_t)(b * kSeq + s)) * kKVDim + h * kHD + 2 * j;
    _Float16* ov = vt + (((size_t)(b * kNKV + h)) * kHD + 2 * j) * kSeq + s;
    ov[0]    = (_Float16)pv[0];
    ov[kSeq] = (_Float16)pv[1];
  }
}

// ---------------------------------------------------------------------------
// Flash attention, one wave32 per (b, head, 16-query tile). All operands f16
// in fragment-native layouts -> pure b128 fragment loads. Key blocks of 32 so
// every WMMA runs at full K depth. Online softmax with 16-lane shuffle
// reductions. Output f16 [b][s][h*64+d] feeds the O-projection GEMM directly.
// ---------------------------------------------------------------------------
__global__ __launch_bounds__(32)
void attn_fa_wmma(const _Float16* __restrict__ qr,
                  const _Float16* __restrict__ kr,
                  const _Float16* __restrict__ vt,
                  _Float16* __restrict__ attn) {
  __shared__ __align__(16) _Float16 Psh[16][40];  // P: C-layout -> A-layout
  const int lane = threadIdx.x;
  const int row  = lane & 15;
  const int kh   = lane >> 4;
  const int nQT  = kSeq / 16;
  int bid = blockIdx.x;
  int qt  = bid % nQT;
  int h   = (bid / nQT) % kNH;
  int b   = bid / (nQT * kNH);
  int kvh = h >> 2;  // GQA: 4 query heads share a KV head
  int q0  = qt * 16;

  const _Float16* Qp = qr + (((size_t)(b * kNH + h)) * kSeq + q0) * kHD;
  const _Float16* Kp = kr + ((size_t)(b * kNKV + kvh)) * kSeq * kHD;
  const _Float16* Vp = vt + ((size_t)(b * kNKV + kvh)) * kHD * kSeq;

  // Q fragments (loop invariant), head-dim chunks of 32
  v16h aQ[2];
#pragma unroll
  for (int ch = 0; ch < 2; ++ch) {
    const _Float16* base = Qp + (size_t)row * kHD + ch * 32;
    aQ[ch] = cat8(*reinterpret_cast<const v8h*>(base + kh * 16),
                  *reinterpret_cast<const v8h*>(base + kh * 16 + 8));
  }

  v8f O[4] = {};
  float mrow[8], lrow[8];
#pragma unroll
  for (int r = 0; r < 8; ++r) { mrow[r] = -1e30f; lrow[r] = 0.0f; }
  const float scale = 0.125f;  // 1/sqrt(64)

  for (int kb = 0; kb <= q0 + 15; kb += 32) {
    // K^T fragments: element e <-> dim 16*kh+e (+ch*32), col = key
    v16h bK[2][2];
#pragma unroll
    for (int nt = 0; nt < 2; ++nt) {
      const _Float16* kbase = Kp + (size_t)(kb + nt * 16 + row) * kHD;
#pragma unroll
      for (int ch = 0; ch < 2; ++ch) {
        const _Float16* p = kbase + ch * 32 + kh * 16;
        bK[nt][ch] = cat8(*reinterpret_cast<const v8h*>(p),
                          *reinterpret_cast<const v8h*>(p + 8));
      }
    }
    v8f s0 = {}, s1 = {};
    s0 = wmma32(aQ[0], bK[0][0], s0);
    s0 = wmma32(aQ[1], bK[0][1], s0);
    s1 = wmma32(aQ[0], bK[1][0], s1);
    s1 = wmma32(aQ[1], bK[1][1], s1);

    // scale + causal mask + online softmax (row = q0+kh*8+r, col = kb(+16)+row)
#pragma unroll
    for (int r = 0; r < 8; ++r) {
      int qi = q0 + kh * 8 + r;
      s0[r] = (kb + row > qi) ? -1e30f : s0[r] * scale;
      s1[r] = (kb + 16 + row > qi) ? -1e30f : s1[r] * scale;
      float mx = fmaxf(s0[r], s1[r]);
#pragma unroll
      for (int off = 8; off >= 1; off >>= 1)
        mx = fmaxf(mx, __shfl_xor(mx, off));
      float mn   = fmaxf(mrow[r], mx);
      float corr = __expf(mrow[r] - mn);
      mrow[r]    = mn;
      float p0   = __expf(s0[r] - mn);
      float p1   = __expf(s1[r] - mn);
      s0[r] = p0;
      s1[r] = p1;
      float ps = p0 + p1;
#pragma unroll
      for (int off = 8; off >= 1; off >>= 1)
        ps += __shfl_xor(ps, off);
      lrow[r] = lrow[r] * corr + ps;
      O[0][r] *= corr; O[1][r] *= corr; O[2][r] *= corr; O[3][r] *= corr;
    }

    // C-layout -> A-layout conversion of P through LDS (f16)
    __syncthreads();
#pragma unroll
    for (int r = 0; r < 8; ++r) {
      Psh[kh * 8 + r][row]      = (_Float16)s0[r];
      Psh[kh * 8 + r][16 + row] = (_Float16)s1[r];
    }
    __syncthreads();
    v16h Pf = cat8(*reinterpret_cast<const v8h*>(&Psh[row][kh * 16]),
                   *reinterpret_cast<const v8h*>(&Psh[row][kh * 16 + 8]));

    // PV: V transposed [d][s] so element e <-> key kb+16*kh+e is contiguous
#pragma unroll
    for (int c = 0; c < 4; ++c) {
      const _Float16* vb = Vp + (size_t)(c * 16 + row) * kSeq + kb + kh * 16;
      v16h bV = cat8(*reinterpret_cast<const v8h*>(vb),
                     *reinterpret_cast<const v8h*>(vb + 8));
      O[c] = wmma32(Pf, bV, O[c]);
    }
  }

  // Normalize, emit f16 [b][s][h*64+d] for the output projection GEMM
#pragma unroll
  for (int r = 0; r < 8; ++r) {
    float inv   = 1.0f / lrow[r];
    size_t orow = (size_t)b * kSeq + q0 + kh * 8 + r;
#pragma unroll
    for (int c = 0; c < 4; ++c)
      attn[orow * kDim + h * kHD + c * 16 + row] = (_Float16)(O[c][r] * inv);
  }
}

// ---------------------------------------------------------------------------
// Host-side orchestration (all on `stream`, graph-capture safe)
// ---------------------------------------------------------------------------
extern "C" void kernel_launch(void* const* d_in, const int* in_sizes, int n_in,
                              void* d_out, int out_size, void* d_ws,
                              size_t ws_size, hipStream_t stream) {
  (void)in_sizes; (void)n_in; (void)out_size; (void)ws_size;
  const float* x    = (const float*)d_in[0];
  const float* cosT = (const float*)d_in[1];
  const float* sinT = (const float*)d_in[2];
  const float* wq   = (const float*)d_in[3];
  const float* wk   = (const float*)d_in[4];
  const float* wv   = (const float*)d_in[5];
  const float* wo   = (const float*)d_in[6];
  float* out = (float*)d_out;

  // Workspace layout (~108 MB). attn16 reuses the qf f32 region.
  char* p = (char*)d_ws;
  _Float16* x16  = (_Float16*)p; p += (size_t)kM * kDim * 2;
  _Float16* wqT  = (_Float16*)p; p += (size_t)kDim * kDim * 2;
  _Float16* wkT  = (_Float16*)p; p += (size_t)kKVDim * kDim * 2;
  _Float16* wvT  = (_Float16*)p; p += (size_t)kKVDim * kDim * 2;
  _Float16* woT  = (_Float16*)p; p += (size_t)kDim * kDim * 2;
  float*    qf   = (float*)p;    p += (size_t)kM * kDim * 4;
  float*    kf   = (float*)p;    p += (size_t)kM * kKVDim * 4;
  float*    vf   = (float*)p;    p += (size_t)kM * kKVDim * 4;
  _Float16* qr16 = (_Float16*)p; p += (size_t)kM * kDim * 2;
  _Float16* kr16 = (_Float16*)p; p += (size_t)kM * kKVDim * 2;
  _Float16* vt16 = (_Float16*)p; p += (size_t)kM * kKVDim * 2;
  _Float16* attn16 = (_Float16*)qf;  // 16 MB inside the 32 MB qf region

  // One-time f16 conversion / weight transposition
  int n4x = kM * kDim / 4;
  cvt_f32_f16<<<(n4x + 255) / 256, 256, 0, stream>>>(x, x16, n4x);
  transpose_f32_f16<<<dim3(kDim / 32, kDim / 32), dim3(32, 8), 0, stream>>>(
      wq, wqT, kDim, kDim);
  transpose_f32_f16<<<dim3(kKVDim / 32, kDim / 32), dim3(32, 8), 0, stream>>>(
      wk, wkT, kDim, kKVDim);
  transpose_f32_f16<<<dim3(kKVDim / 32, kDim / 32), dim3(32, 8), 0, stream>>>(
      wv, wvT, kDim, kKVDim);
  transpose_f32_f16<<<dim3(kDim / 32, kDim / 32), dim3(32, 8), 0, stream>>>(
      wo, woT, kDim, kDim);

  // QKV projections (f32 accumulate/output)
  gemm_wmma_f16<<<dim3(kDim / 64, kM / 128), 128, 0, stream>>>(
      x16, wqT, qf, kM, kDim, kDim);
  gemm_wmma_f16<<<dim3(kKVDim / 64, kM / 128), 128, 0, stream>>>(
      x16, wkT, kf, kM, kKVDim, kDim);
  gemm_wmma_f16<<<dim3(kKVDim / 64, kM / 128), 128, 0, stream>>>(
      x16, wvT, vf, kM, kKVDim, kDim);

  // RoPE + relayout to fragment-native f16
  int ropeN = kBsz * kSeq * kNH * 32;
  rope_qkv<<<ropeN / 256, 256, 0, stream>>>(qf, kf, vf, cosT, sinT, qr16, kr16,
                                            vt16);

  // Flash attention (writes f16 into attn16 == qf region)
  attn_fa_wmma<<<kBsz * kNH * (kSeq / 16), 32, 0, stream>>>(qr16, kr16, vt16,
                                                            attn16);

  // Output projection
  gemm_wmma_f16<<<dim3(kDim / 64, kM / 128), 128, 0, stream>>>(
      attn16, woT, out, kM, kDim, kDim);
}
